// GraphSAGENet_30081950941185
// MI455X (gfx1250) — matrix operand order repaired and checked by reading, not compiled, not run on hardware
//
#include <hip/hip_runtime.h>
#include <hip/hip_bf16.h>

// ---------------------------------------------------------------------------
// GraphSAGE (2x SAGEConv + mean-pool + linear head) for MI455X / gfx1250.
// Scatter-mean phases: f32 global atomics (memory-bound, ~2.5GB traffic).
// Dense GEMMs: v_wmma_f32_16x16x32_bf16, wave32, one 16x16 tile per wave,
// K fully unrolled at compile time (4 WMMAs for layer1, 8 for layer2).
// ---------------------------------------------------------------------------

typedef __attribute__((ext_vector_type(16))) __bf16 v16bf;
typedef __attribute__((ext_vector_type(8)))  float  v8f;

#define N_NODES   100000
#define IN_CH     64
#define HID       128
#define N_CLASSES 8
#define N_GRAPHS  1000

// ---- workspace layout (float offsets) -------------------------------------
// zero-region (memset each call): msg1 | cnt1 | msg2 | pool | gcnt
#define F_MSG1  ((size_t)0)
#define F_CNT1  (F_MSG1 + (size_t)N_NODES * IN_CH)
#define F_MSG2  (F_CNT1 + (size_t)N_NODES)
#define F_POOL  (F_MSG2 + (size_t)N_NODES * HID)
#define F_GCNT  (F_POOL + (size_t)N_GRAPHS * HID)
#define F_ZEND  (F_GCNT + (size_t)N_GRAPHS)
#define F_H1    (F_ZEND)
#define F_H2    (F_H1 + (size_t)N_NODES * HID)
#define F_PANEL (F_H2 + (size_t)N_NODES * HID)              // bf16 region
// packed bf16 weight panels (counts in bf16 elements)
#define P_W1L   ((size_t)0)
#define P_W1R   (P_W1L + 8192)     // 8 coltiles * 2 ksteps * 512
#define P_W2L   (P_W1R + 8192)
#define P_W2R   (P_W2L + 16384)    // 8 coltiles * 4 ksteps * 512

// ---------------------------------------------------------------------------
// Pack a [K,128] f32 weight matrix into bf16 panels matching the ISA B-fragment
// layout for V_WMMA_F32_16X16X32_BF16:
//   panel index p = coltile*(K/32) + kkstep ; per panel: 32 lanes x 16 bf16
//   lane l (<16): N = 16*coltile + l,      K = 32*kk + (j&7) + 16*(j>>3)      (j = 0..15)
//   lane l (>=16): N = 16*coltile + l-16,  K = 32*kk + 8 + (j&7) + 16*(j>>3)
// ---------------------------------------------------------------------------
__global__ void sage_pack_b(const float* __restrict__ W, __bf16* __restrict__ dst, int K) {
    int i = blockIdx.x * blockDim.x + threadIdx.x;
    int nkk = K >> 5;
    int total = 8 * nkk * 32 * 16;
    if (i >= total) return;
    int j    = i & 15;
    int lane = (i >> 4) & 31;
    int p    = i >> 9;
    int c    = p / nkk;
    int kk   = p - c * nkk;
    int n    = 16 * c + (lane & 15);
    int k    = 32 * kk + 8 * (lane >> 4) + (j & 7) + 16 * (j >> 3);
    dst[i] = (__bf16)W[k * HID + n];
}

// ---------------------------------------------------------------------------
// Edge scatter: msg[dst] += feat[src] (float4 per thread), optional in-degree
// count. C4 = channels/4 (16 for layer1, 32 for layer2), compile-time so
// index math is shift/mask.
// ---------------------------------------------------------------------------
template <int C4, bool DO_CNT>
__global__ void sage_scatter(const float* __restrict__ feat,
                             const int* __restrict__ src,
                             const int* __restrict__ dst,
                             float* __restrict__ msg,
                             float* __restrict__ cnt,
                             int nedges) {
    int i = blockIdx.x * blockDim.x + threadIdx.x;
    if (i >= nedges * C4) return;
    int e = i / C4;                       // C4 is a power of two -> shift
    int g = i & (C4 - 1);
    int s = src[e];
    int d = dst[e];
    float4 v = ((const float4*)(feat + (size_t)s * C4 * 4))[g];
    float* base = msg + (size_t)d * C4 * 4 + (size_t)g * 4;
    atomicAdd(base + 0, v.x);
    atomicAdd(base + 1, v.y);
    atomicAdd(base + 2, v.z);
    atomicAdd(base + 3, v.w);
    if (DO_CNT && g == 0) atomicAdd(cnt + d, 1.0f);
}

// ---- A-fragment builder ----------------------------------------------------
// ISA 16-bit A 16x32 layout: lane l holds row (l&15);
//   j<8  -> K = kk + 8*(l>>4) + j ; j>=8 -> K = kk + 16 + 8*(l>>4) + (j-8)
__device__ __forceinline__ v16bf a_frag_from_f32(const float* __restrict__ row,
                                                 int base0, float scale) {
    float4 a0 = *(const float4*)(row + base0);
    float4 a1 = *(const float4*)(row + base0 + 4);
    float4 a2 = *(const float4*)(row + base0 + 16);
    float4 a3 = *(const float4*)(row + base0 + 20);
    v16bf r;
    r[0]  = (__bf16)(a0.x * scale); r[1]  = (__bf16)(a0.y * scale);
    r[2]  = (__bf16)(a0.z * scale); r[3]  = (__bf16)(a0.w * scale);
    r[4]  = (__bf16)(a1.x * scale); r[5]  = (__bf16)(a1.y * scale);
    r[6]  = (__bf16)(a1.z * scale); r[7]  = (__bf16)(a1.w * scale);
    r[8]  = (__bf16)(a2.x * scale); r[9]  = (__bf16)(a2.y * scale);
    r[10] = (__bf16)(a2.z * scale); r[11] = (__bf16)(a2.w * scale);
    r[12] = (__bf16)(a3.x * scale); r[13] = (__bf16)(a3.y * scale);
    r[14] = (__bf16)(a3.z * scale); r[15] = (__bf16)(a3.w * scale);
    return r;
}

// ---------------------------------------------------------------------------
// SAGEConv GEMM: out = relu( (msg/max(cnt,1)) @ Wl + feat @ Wr + bias )
// grid.x = N/16 row tiles; block = 256 (8 waves); wave w -> columns [16w,16w+16)
// K compile-time: loops fully unroll into straight-line WMMA chains.
// ---------------------------------------------------------------------------
template <int K>
__global__ void sage_gemm(const float* __restrict__ feat,
                          const float* __restrict__ msg,
                          const float* __restrict__ cnt,
                          const __bf16* __restrict__ Bl,
                          const __bf16* __restrict__ Br,
                          const float* __restrict__ bias,
                          float* __restrict__ out) {
    constexpr int NKK = K >> 5;
    const int lane = threadIdx.x & 31;
    const int wave = threadIdx.x >> 5;          // col tile 0..7
    const int r16  = lane & 15;
    const int kg   = lane >> 4;
    const int node = blockIdx.x * 16 + r16;     // N divisible by 16

    const float* frow = feat + (size_t)node * K;
    const float* mrow = msg  + (size_t)node * K;
    const float  invc = 1.0f / fmaxf(cnt[node], 1.0f);

    const __bf16* panelL = Bl + (size_t)wave * NKK * 512 + (size_t)lane * 16;
    const __bf16* panelR = Br + (size_t)wave * NKK * 512 + (size_t)lane * 16;

    v8f acc = {};

    // mean-aggregated neighbor term: (msg * invc) @ Wl
    #pragma unroll
    for (int kk = 0; kk < NKK; ++kk) {
        v16bf a = a_frag_from_f32(mrow, kk * 32 + 8 * kg, invc);
        v16bf b = *(const v16bf*)(panelL + (size_t)kk * 512);
        acc = __builtin_amdgcn_wmma_f32_16x16x32_bf16(false, a, false, b,
                                                      (short)0, acc, false, false);
    }
    // self term: feat @ Wr
    #pragma unroll
    for (int kk = 0; kk < NKK; ++kk) {
        v16bf a = a_frag_from_f32(frow, kk * 32 + 8 * kg, 1.0f);
        v16bf b = *(const v16bf*)(panelR + (size_t)kk * 512);
        acc = __builtin_amdgcn_wmma_f32_16x16x32_bf16(false, a, false, b,
                                                      (short)0, acc, false, false);
    }

    // C/D layout: acc[r] -> row (r + 8*kg), col (lane&15) of the 16x16 tile
    const int n  = wave * 16 + r16;
    const float bz = bias[n];
    #pragma unroll
    for (int r = 0; r < 8; ++r) {
        float v = fmaxf(acc[r] + bz, 0.0f);
        out[(size_t)(blockIdx.x * 16 + r + 8 * kg) * HID + n] = v;
    }
}

// ---------------------------------------------------------------------------
// Per-graph mean pool scatter: pool[batch[n]] += h2[n]; gcnt[batch[n]] += 1
// ---------------------------------------------------------------------------
__global__ void sage_pool(const float* __restrict__ h2,
                          const int* __restrict__ batch,
                          float* __restrict__ pool,
                          float* __restrict__ gcnt) {
    int i = blockIdx.x * blockDim.x + threadIdx.x;   // N * 32 float4 groups
    if (i >= N_NODES * 32) return;
    int n  = i >> 5;
    int g4 = i & 31;
    int g  = batch[n];
    float4 v = ((const float4*)(h2 + (size_t)n * HID))[g4];
    float* base = pool + (size_t)g * HID + (size_t)g4 * 4;
    atomicAdd(base + 0, v.x);
    atomicAdd(base + 1, v.y);
    atomicAdd(base + 2, v.z);
    atomicAdd(base + 3, v.w);
    if (g4 == 0) atomicAdd(gcnt + g, 1.0f);
}

// ---------------------------------------------------------------------------
// Head: out[g,c] = sum_k (pool[g,k]/max(gcnt,1)) * Wh[k,c] + bh[c]
// ---------------------------------------------------------------------------
__global__ void sage_head(const float* __restrict__ pool,
                          const float* __restrict__ gcnt,
                          const float* __restrict__ Wh,
                          const float* __restrict__ bh,
                          float* __restrict__ out) {
    int i = blockIdx.x * blockDim.x + threadIdx.x;
    if (i >= N_GRAPHS * N_CLASSES) return;
    int g = i >> 3;
    int c = i & 7;
    float invc = 1.0f / fmaxf(gcnt[g], 1.0f);
    float s = bh[c];
    const float* prow = pool + (size_t)g * HID;
    #pragma unroll 8
    for (int k = 0; k < HID; ++k) s += prow[k] * invc * Wh[k * N_CLASSES + c];
    out[i] = s;
}

extern "C" void kernel_launch(void* const* d_in, const int* in_sizes, int n_in,
                              void* d_out, int out_size, void* d_ws, size_t ws_size,
                              hipStream_t stream) {
    const float* x    = (const float*)d_in[0];
    const int*   eidx = (const int*)d_in[1];
    const int*   batch= (const int*)d_in[2];
    const float* W1l  = (const float*)d_in[3];
    const float* W1r  = (const float*)d_in[4];
    const float* b1   = (const float*)d_in[5];
    const float* W2l  = (const float*)d_in[6];
    const float* W2r  = (const float*)d_in[7];
    const float* b2   = (const float*)d_in[8];
    const float* Wh   = (const float*)d_in[9];
    const float* bh   = (const float*)d_in[10];
    float* out = (float*)d_out;

    const int E = in_sizes[1] / 2;
    const int* src = eidx;
    const int* dst = eidx + E;

    float*  ws    = (float*)d_ws;
    float*  msg1  = ws + F_MSG1;
    float*  cnt1  = ws + F_CNT1;
    float*  msg2  = ws + F_MSG2;
    float*  pool  = ws + F_POOL;
    float*  gcnt  = ws + F_GCNT;
    float*  h1    = ws + F_H1;
    float*  h2    = ws + F_H2;
    __bf16* panel = (__bf16*)(ws + F_PANEL);
    __bf16* p1l = panel + P_W1L;
    __bf16* p1r = panel + P_W1R;
    __bf16* p2l = panel + P_W2L;
    __bf16* p2r = panel + P_W2R;

    // zero all atomic-accumulated buffers (contiguous region) — capture-safe
    hipMemsetAsync(ws, 0, F_ZEND * sizeof(float), stream);

    // pack weights into WMMA B-fragment bf16 panels
    sage_pack_b<<<(8192 + 255) / 256, 256, 0, stream>>>(W1l, p1l, IN_CH);
    sage_pack_b<<<(8192 + 255) / 256, 256, 0, stream>>>(W1r, p1r, IN_CH);
    sage_pack_b<<<(16384 + 255) / 256, 256, 0, stream>>>(W2l, p2l, HID);
    sage_pack_b<<<(16384 + 255) / 256, 256, 0, stream>>>(W2r, p2r, HID);

    // layer 1: scatter-mean over edges (64 ch) + counts, then WMMA GEMM
    {
        int total = E * (IN_CH / 4);
        sage_scatter<IN_CH / 4, true><<<(total + 255) / 256, 256, 0, stream>>>(
            x, src, dst, msg1, cnt1, E);
        sage_gemm<IN_CH><<<N_NODES / 16, 256, 0, stream>>>(
            x, msg1, cnt1, p1l, p1r, b1, h1);
    }
    // layer 2: scatter-mean over edges (128 ch), then WMMA GEMM
    {
        int total = E * (HID / 4);
        sage_scatter<HID / 4, false><<<(total + 255) / 256, 256, 0, stream>>>(
            h1, src, dst, msg2, cnt1, E);
        sage_gemm<HID><<<N_NODES / 16, 256, 0, stream>>>(
            h1, msg2, cnt1, p2l, p2r, b2, h2);
    }
    // global mean pool + linear head
    sage_pool<<<(N_NODES * 32 + 255) / 256, 256, 0, stream>>>(h2, batch, pool, gcnt);
    sage_head<<<(N_GRAPHS * N_CLASSES + 255) / 256, 256, 0, stream>>>(
        pool, gcnt, Wh, bh, out);
}
